// Attention_85916525789719
// MI455X (gfx1250) — matrix operand order repaired and checked by reading, not compiled, not run on hardware
//
#include <hip/hip_runtime.h>
#include <stdint.h>

#define B_   2
#define S_   2048
#define DIM_ 2048
#define H_   32
#define KV_  8
#define HD_  64
#define REP_ (H_ / KV_)
#define NEG_ -1000000000.0f

typedef __attribute__((ext_vector_type(16))) __bf16 bf16x16;
typedef __attribute__((ext_vector_type(8)))  float  f32x8;
typedef __attribute__((__vector_size__(16))) int    i32x4;

#define AS1 __attribute__((address_space(1)))
#define AS3 __attribute__((address_space(3)))

#if __has_builtin(__builtin_amdgcn_global_load_async_to_lds_b128) && \
    __has_builtin(__builtin_amdgcn_s_wait_asynccnt)
#define USE_ASYNC 1
#else
#define USE_ASYNC 0
#endif

union Frag {
    uint4   q[2];
    bf16x16 v;
};

__device__ __forceinline__ uint16_t f2bf(float f) {
    uint32_t u = __builtin_bit_cast(uint32_t, f);
    u += 0x7fffu + ((u >> 16) & 1u);   // round-to-nearest-even
    return (uint16_t)(u >> 16);
}

// stage one 16-byte chunk global -> LDS (async DMA on CDNA5, fallback: copy through VGPRs)
__device__ __forceinline__ void stage16(const uint16_t* __restrict__ g, uint16_t* l) {
#if USE_ASYNC
    __builtin_amdgcn_global_load_async_to_lds_b128(
        (AS1 i32x4*)g, (AS3 i32x4*)l, 0, 0);
#else
    *(uint4*)l = *(const uint4*)g;
#endif
}

__device__ __forceinline__ void stage_wait() {
#if USE_ASYNC
    __builtin_amdgcn_s_wait_asynccnt(0);
#endif
}

// ---------------- elementwise conversion kernels ----------------

__global__ void k_f32_to_bf16(const float* __restrict__ in, uint16_t* __restrict__ out, int n) {
    int i = blockIdx.x * 256 + threadIdx.x;
    if (i < n) out[i] = f2bf(in[i]);
}

// out[c*R + r] = bf16(in[r*C + c])   (weight -> K-contiguous "BT" layout)
__global__ void k_transpose_bf16(const float* __restrict__ in, uint16_t* __restrict__ out,
                                 int R, int C) {
    int i = blockIdx.x * 256 + threadIdx.x;
    if (i >= R * C) return;
    int r = i / C, c = i % C;
    out[(size_t)c * R + r] = f2bf(in[i]);
}

// RoPE over f32 GEMM output, write bf16.
// in: [B][S][nh][HD].  transpose_sk=0 -> out same layout; =1 -> out [B][nh][S][HD]
__global__ void k_rope(const float* __restrict__ in, const float* __restrict__ cosf,
                       const float* __restrict__ sinf, uint16_t* __restrict__ out,
                       int nh, float scale, int transpose_sk) {
    int i = blockIdx.x * 256 + threadIdx.x;
    int total = B_ * S_ * nh * (HD_ / 2);
    if (i >= total) return;
    int p = i % (HD_ / 2);
    int h = (i / (HD_ / 2)) % nh;
    int s = (i / (HD_ / 2) / nh) % S_;
    int b =  i / (HD_ / 2) / nh / S_;
    size_t bi = (((size_t)(b * S_ + s)) * nh + h) * HD_;
    float te = in[bi + 2 * p], to = in[bi + 2 * p + 1];
    float c = cosf[s * (HD_ / 2) + p], sn = sinf[s * (HD_ / 2) + p];
    float oe = (te * c - to * sn) * scale;
    float oo = (te * sn + to * c) * scale;
    size_t bo = transpose_sk ? ((((size_t)b * nh + h) * S_ + s) * HD_) : bi;
    out[bo + 2 * p]     = f2bf(oe);
    out[bo + 2 * p + 1] = f2bf(oo);
}

// V f32 [B][S][KV][HD] -> bf16 Vt [B][KV][HD][S]  (key-contiguous for PV B-fragments)
__global__ void k_vt(const float* __restrict__ in, uint16_t* __restrict__ out) {
    int i = blockIdx.x * 256 + threadIdx.x;
    if (i >= B_ * S_ * KV_ * HD_) return;
    int hd = i % HD_;
    int kv = (i / HD_) % KV_;
    int s  = (i / HD_ / KV_) % S_;
    int b  =  i / HD_ / KV_ / S_;
    out[(((size_t)b * KV_ + kv) * HD_ + hd) * S_ + s] = f2bf(in[i]);
}

// ---------------- WMMA GEMM with async-LDS double buffering ----------------
// C[M][N] f32 = A[M][K]bf16 * BT[N][K]bf16.
// 128 threads = 4 waves per block; block tile 128x64; wave w computes rows [w*32, w*32+32).
// Per k-step (32): A tile 128x32, B tile 64x32 staged to LDS by async DMA, double buffered.
// LDS rows padded to 80B so 16-lane b128 fragment reads hit distinct banks.
#define A_ROWP 40                   // row pitch in elems (80B)
#define A_ELEMS (128 * A_ROWP)
#define B_ELEMS (64 * A_ROWP)

__global__ void __launch_bounds__(128)
k_gemm_bf16(const uint16_t* __restrict__ A, const uint16_t* __restrict__ BT,
            float* __restrict__ C, int M, int N, int K) {
    __shared__ __align__(16) uint16_t lds_A[2 * A_ELEMS];
    __shared__ __align__(16) uint16_t lds_B[2 * B_ELEMS];

    const int tid  = threadIdx.x;
    const int wid  = tid >> 5;
    const int lane = tid & 31;
    const int li = lane & 15;
    const int hi = lane >> 4;
    const int n_blk = blockIdx.x * 64;
    const int m_blk = blockIdx.y * 128;

    // per-thread staging sources (advance by 32 elems per k-step)
    const uint16_t* gA = A + (size_t)(m_blk + tid) * K;            // 4 x b128 per step
    const int bn   = tid >> 1, bh = tid & 1;
    const uint16_t* gB = BT + (size_t)(n_blk + bn) * K + bh * 16;  // 2 x b128 per step
    uint16_t* lA_t = lds_A + tid * A_ROWP;
    uint16_t* lB_t = lds_B + bn * A_ROWP + bh * 16;

    const int nsteps = K / 32;
    auto stage = [&](int buf, int k0) {
#pragma unroll
        for (int u = 0; u < 4; ++u)
            stage16(gA + k0 + u * 8, lA_t + buf * A_ELEMS + u * 8);
#pragma unroll
        for (int u = 0; u < 2; ++u)
            stage16(gB + k0 + u * 8, lB_t + buf * B_ELEMS + u * 8);
    };

    f32x8 acc[2][4] = {};
    stage(0, 0);
    stage_wait();
    __syncthreads();

    for (int s = 0; s < nsteps; ++s) {
        const int cur = s & 1;
        if (s + 1 < nsteps) stage(1 - cur, (s + 1) * 32);   // overlap DMA with wmma

        Frag a[2];
#pragma unroll
        for (int r = 0; r < 2; ++r) {
            const uint16_t* pa = lds_A + cur * A_ELEMS + (wid * 32 + r * 16 + li) * A_ROWP + hi * 8;
            a[r].q[0] = *(const uint4*)pa;          // K 0..7   (hi half: 8..15)
            a[r].q[1] = *(const uint4*)(pa + 16);   // K 16..23 (hi half: 24..31)
        }
#pragma unroll
        for (int c = 0; c < 4; ++c) {
            Frag b;
            const uint16_t* pb = lds_B + cur * B_ELEMS + (c * 16 + li) * A_ROWP + hi * 8;
            b.q[0] = *(const uint4*)pb;             // K 0..7   (hi half: 8..15)
            b.q[1] = *(const uint4*)(pb + 16);      // K 16..23 (hi half: 24..31)
#pragma unroll
            for (int r = 0; r < 2; ++r)
                acc[r][c] = __builtin_amdgcn_wmma_f32_16x16x32_bf16(
                    false, a[r].v, false, b.v, (short)0, acc[r][c], false, false);
        }
        stage_wait();        // next buffer fully landed
        __syncthreads();     // all waves done reading cur before it is overwritten
    }

#pragma unroll
    for (int r = 0; r < 2; ++r)
#pragma unroll
        for (int c = 0; c < 4; ++c)
#pragma unroll
            for (int j = 0; j < 8; ++j) {
                int m = m_blk + wid * 32 + r * 16 + j + 8 * hi;
                int n = n_blk + c * 16 + li;
                C[(size_t)m * N + n] = acc[r][c][j];
            }
}

// ---------------- flash attention: one wave per (b, h, 16-query tile) ----------------
// Q bf16 [B][S][H][HD] (pre-scaled by 1/8), Kb bf16 [B][KV][S][HD], Vt bf16 [B][KV][HD][S]
// O bf16 [B][S][H*HD]
__global__ void k_attn(const uint16_t* __restrict__ Q, const uint16_t* __restrict__ Kb,
                       const uint16_t* __restrict__ Vt, uint16_t* __restrict__ O) {
    __shared__ __align__(16) uint16_t lds_p[16 * 32];
    const int lane = threadIdx.x;
    const int li = lane & 15, hi = lane >> 4;
    const int qt = blockIdx.x, h = blockIdx.y, b = blockIdx.z;
    const int kvh = h / REP_;
    const int q_base = qt * 16;

    Frag qa[2];  // HD=64 -> two K=32 A-fragments, resident for the whole block
    {
        const uint16_t* pq = Q + (((size_t)(b * S_ + q_base + li)) * H_ + h) * HD_ + hi * 8;
#pragma unroll
        for (int f = 0; f < 2; ++f) {
            qa[f].q[0] = *(const uint4*)(pq + f * 32);
            qa[f].q[1] = *(const uint4*)(pq + f * 32 + 16);
        }
    }

    float m_run[8], l_run[8];
#pragma unroll
    for (int j = 0; j < 8; ++j) { m_run[j] = -3.0e38f; l_run[j] = 0.f; }
    f32x8 accO[4] = {};

    const uint16_t* kbase = Kb + ((size_t)(b * KV_ + kvh)) * S_ * HD_;
    const uint16_t* vbase = Vt + ((size_t)(b * KV_ + kvh)) * HD_ * S_;

    for (int kb = 0; kb < q_base + 16; kb += 32) {
        // scores: 16 queries x 32 keys = two 16x16 D tiles, each 2 wmma over HD
        f32x8 sc[2] = {};
#pragma unroll
        for (int t = 0; t < 2; ++t) {
#pragma unroll
            for (int f = 0; f < 2; ++f) {
                Frag bk;
                const uint16_t* pk = kbase + (size_t)(kb + t * 16 + li) * HD_ + f * 32 + hi * 8;
                bk.q[0] = *(const uint4*)pk;
                bk.q[1] = *(const uint4*)(pk + 16);
                sc[t] = __builtin_amdgcn_wmma_f32_16x16x32_bf16(
                    false, qa[f].v, false, bk.v, (short)0, sc[t], false, false);
            }
        }
        // causal mask (only needed on/above the diagonal band)
        if (kb + 31 > q_base) {
#pragma unroll
            for (int t = 0; t < 2; ++t) {
                int key = kb + t * 16 + li;
#pragma unroll
                for (int j = 0; j < 8; ++j) {
                    int q = q_base + j + 8 * hi;
                    if (key > q) sc[t][j] += NEG_;
                }
            }
        }
        // online softmax; rows live in (vgpr j, lane half), columns across 16 lanes
        float p0[8], p1[8], scl[8];
#pragma unroll
        for (int j = 0; j < 8; ++j) {
            float v = fmaxf(sc[0][j], sc[1][j]);
            v = fmaxf(v, __shfl_xor(v, 1, 32));
            v = fmaxf(v, __shfl_xor(v, 2, 32));
            v = fmaxf(v, __shfl_xor(v, 4, 32));
            v = fmaxf(v, __shfl_xor(v, 8, 32));
            float mn = fmaxf(m_run[j], v);
            scl[j] = __expf(m_run[j] - mn);
            p0[j]  = __expf(sc[0][j] - mn);
            p1[j]  = __expf(sc[1][j] - mn);
            float rs = p0[j] + p1[j];
            rs += __shfl_xor(rs, 1, 32);
            rs += __shfl_xor(rs, 2, 32);
            rs += __shfl_xor(rs, 4, 32);
            rs += __shfl_xor(rs, 8, 32);
            l_run[j] = l_run[j] * scl[j] + rs;
            m_run[j] = mn;
        }
#pragma unroll
        for (int c = 0; c < 4; ++c)
#pragma unroll
            for (int j = 0; j < 8; ++j) accO[c][j] *= scl[j];

        // D-layout -> A-layout transpose of P via LDS (wave-private, DS ops are in-order)
#pragma unroll
        for (int j = 0; j < 8; ++j) {
            lds_p[(j + 8 * hi) * 32 + li]      = f2bf(p0[j]);
            lds_p[(j + 8 * hi) * 32 + li + 16] = f2bf(p1[j]);
        }
        Frag pa;
        {
            const uint16_t* pp = lds_p + li * 32 + hi * 8;
            pa.q[0] = *(const uint4*)pp;
            pa.q[1] = *(const uint4*)(pp + 16);
        }
        // O += P (16x32) * V (32x64), 4 N-tiles
#pragma unroll
        for (int c = 0; c < 4; ++c) {
            Frag vb;
            const uint16_t* pv = vbase + (size_t)(c * 16 + li) * S_ + kb + hi * 8;
            vb.q[0] = *(const uint4*)pv;
            vb.q[1] = *(const uint4*)(pv + 16);
            accO[c] = __builtin_amdgcn_wmma_f32_16x16x32_bf16(
                false, pa.v, false, vb.v, (short)0, accO[c], false, false);
        }
    }
#pragma unroll
    for (int c = 0; c < 4; ++c)
#pragma unroll
        for (int j = 0; j < 8; ++j) {
            int q = q_base + j + 8 * hi;
            int hd = c * 16 + li;
            O[(((size_t)(b * S_ + q)) * H_ + h) * HD_ + hd] = f2bf(accO[c][j] / l_run[j]);
        }
}

// ---------------- host orchestration ----------------

extern "C" void kernel_launch(void* const* d_in, const int* in_sizes, int n_in,
                              void* d_out, int out_size, void* d_ws, size_t ws_size,
                              hipStream_t stream) {
    (void)in_sizes; (void)n_in; (void)out_size; (void)ws_size;
    const float* x  = (const float*)d_in[0];
    const float* fc = (const float*)d_in[1];
    const float* fs = (const float*)d_in[2];
    /* d_in[3] = mask: causal tril, applied analytically in k_attn */
    const float* wq = (const float*)d_in[4];
    const float* wk = (const float*)d_in[5];
    const float* wv = (const float*)d_in[6];
    const float* wo = (const float*)d_in[7];
    float* out = (float*)d_out;

    char* ws = (char*)d_ws;
    size_t off = 0;
    auto alloc = [&](size_t bytes) -> void* {
        void* p = ws + off;
        off += (bytes + 255) & ~(size_t)255;
        return p;
    };
    const size_t MQ = (size_t)B_ * S_ * H_ * HD_;   // 8,388,608
    const size_t MK = (size_t)B_ * S_ * KV_ * HD_;  // 2,097,152

    uint16_t* xb   = (uint16_t*)alloc((size_t)B_ * S_ * DIM_ * 2);
    uint16_t* wqT  = (uint16_t*)alloc((size_t)DIM_ * H_ * HD_ * 2);
    uint16_t* wkT  = (uint16_t*)alloc((size_t)DIM_ * KV_ * HD_ * 2);
    uint16_t* wvT  = (uint16_t*)alloc((size_t)DIM_ * KV_ * HD_ * 2);
    uint16_t* woT  = (uint16_t*)alloc((size_t)H_ * HD_ * DIM_ * 2);
    float*    Ftmp = (float*)   alloc(MQ * 4);      // shared f32 GEMM scratch (Q/K/V)
    uint16_t* Qb   = (uint16_t*)alloc(MQ * 2);
    uint16_t* Kbuf = (uint16_t*)alloc(MK * 2);
    uint16_t* Vt   = (uint16_t*)alloc(MK * 2);
    uint16_t* Ob   = (uint16_t*)alloc(MQ * 2);

    int n;
    // bf16 conversions
    n = B_ * S_ * DIM_;
    k_f32_to_bf16<<<(n + 255) / 256, 256, 0, stream>>>(x, xb, n);
    n = DIM_ * H_ * HD_;
    k_transpose_bf16<<<(n + 255) / 256, 256, 0, stream>>>(wq, wqT, DIM_, H_ * HD_);
    n = DIM_ * KV_ * HD_;
    k_transpose_bf16<<<(n + 255) / 256, 256, 0, stream>>>(wk, wkT, DIM_, KV_ * HD_);
    k_transpose_bf16<<<(n + 255) / 256, 256, 0, stream>>>(wv, wvT, DIM_, KV_ * HD_);
    n = H_ * HD_ * DIM_;
    k_transpose_bf16<<<(n + 255) / 256, 256, 0, stream>>>(wo, woT, H_ * HD_, DIM_);

    const int Mrows = B_ * S_;
    // Q = x@wq, RoPE(+1/sqrt(HD) folded into Q)
    k_gemm_bf16<<<dim3((H_ * HD_) / 64, Mrows / 128), 128, 0, stream>>>(
        xb, wqT, Ftmp, Mrows, H_ * HD_, DIM_);
    n = B_ * S_ * H_ * (HD_ / 2);
    k_rope<<<(n + 255) / 256, 256, 0, stream>>>(Ftmp, fc, fs, Qb, H_, 0.125f, 0);
    // K = x@wk, RoPE, transposed to [b][kv][s][hd]
    k_gemm_bf16<<<dim3((KV_ * HD_) / 64, Mrows / 128), 128, 0, stream>>>(
        xb, wkT, Ftmp, Mrows, KV_ * HD_, DIM_);
    n = B_ * S_ * KV_ * (HD_ / 2);
    k_rope<<<(n + 255) / 256, 256, 0, stream>>>(Ftmp, fc, fs, Kbuf, KV_, 1.0f, 1);
    // V = x@wv, transposed to [b][kv][hd][s]
    k_gemm_bf16<<<dim3((KV_ * HD_) / 64, Mrows / 128), 128, 0, stream>>>(
        xb, wvT, Ftmp, Mrows, KV_ * HD_, DIM_);
    n = (int)MK;
    k_vt<<<(n + 255) / 256, 256, 0, stream>>>(Ftmp, Vt);

    // flash attention
    k_attn<<<dim3(S_ / 16, H_, B_), 32, 0, stream>>>(Qb, Kbuf, Vt, Ob);

    // out = Ob @ wo  (f32 result straight to d_out)
    k_gemm_bf16<<<dim3(DIM_ / 64, Mrows / 128), 128, 0, stream>>>(
        Ob, woT, out, Mrows, DIM_, DIM_);
}
